// BlockGNN_65970697666563
// MI455X (gfx1250) — compile-verified, matrix operand order checked by reading, not run on hardware
//
#include <hip/hip_runtime.h>
#include <hip/hip_bf16.h>

typedef __attribute__((ext_vector_type(16))) __bf16 v16bf;
typedef __attribute__((ext_vector_type(2)))  __bf16 v2bf;
typedef __attribute__((ext_vector_type(8)))  float  v8f;

#define HDIM 256

static __device__ __forceinline__ __bf16 f2bf(float f) {
    unsigned int u = __float_as_uint(f);
    u += 0x7FFFu + ((u >> 16) & 1u);           // round-to-nearest-even
    unsigned short s = (unsigned short)(u >> 16);
    return __builtin_bit_cast(__bf16, s);
}

// Packed f32x2 -> bf16x2 conversion: single v_cvt_pk_bf16_f32 when available.
static __device__ __forceinline__ v2bf f2bf_pk(float lo, float hi) {
#if __has_builtin(__builtin_amdgcn_cvt_pk_bf16_f32)
    return __builtin_amdgcn_cvt_pk_bf16_f32(lo, hi);
#else
    v2bf r; r[0] = f2bf(lo); r[1] = f2bf(hi); return r;
#endif
}

// ---------------- degree / norm ----------------
__global__ void k_fill1(float* p, int n) {
    int t = blockIdx.x * blockDim.x + threadIdx.x;
    if (t < n) p[t] = 1.0f;                    // self-loop contributes 1 to every degree
}
__global__ void k_degedge(float* deg, const int* __restrict__ dst, int nE) {
    int t = blockIdx.x * blockDim.x + threadIdx.x;
    if (t < nE) unsafeAtomicAdd(&deg[dst[t]], 1.0f);
}
__global__ void k_rsqrt(float* p, int n) {
    int t = blockIdx.x * blockDim.x + threadIdx.x;
    if (t < n) p[t] = rsqrtf(p[t]);            // deg >= 1 always (self loops)
}

// ---------------- aggregation: out = A_norm @ in ----------------
// self-loop term doubles as the initializer (no pre-zero needed)
__global__ void k_aggself(const float* __restrict__ in, float* __restrict__ out,
                          const float* __restrict__ inv, int D4, long long total) {
    long long t = (long long)blockIdx.x * blockDim.x + threadIdx.x;
    if (t >= total) return;
    int i = (int)(t / D4);
    float w = inv[i] * inv[i];
    float4 v = reinterpret_cast<const float4*>(in)[t];
    float4 o; o.x = v.x * w; o.y = v.y * w; o.z = v.z * w; o.w = v.w * w;
    reinterpret_cast<float4*>(out)[t] = o;
}
__global__ void k_aggedge(const float* __restrict__ in, float* __restrict__ out,
                          const float* __restrict__ inv,
                          const int* __restrict__ src, const int* __restrict__ dst,
                          int D4, long long total) {
    long long t = (long long)blockIdx.x * blockDim.x + threadIdx.x;
    if (t >= total) return;
    int e = (int)(t / D4);
    int c = (int)(t % D4);
    int s = src[e], d = dst[e];
    float w = inv[s] * inv[d];
    float4 v = reinterpret_cast<const float4*>(in)[(long long)s * D4 + c];
    float* o = out + ((long long)d * D4 + c) * 4;
    unsafeAtomicAdd(o + 0, v.x * w);
    unsafeAtomicAdd(o + 1, v.y * w);
    unsafeAtomicAdd(o + 2, v.z * w);
    unsafeAtomicAdd(o + 3, v.w * w);
}

// ---------------- weight pre-pack into CDNA5 WMMA B-operand layout ----------------
// packed index: (((ntile*(K/32) + kc)*32 + lane)*16 + e), 16 bf16 per lane contiguous
__global__ void k_packw(const float* __restrict__ W, __bf16* __restrict__ Wp, int K) {
    int t = blockIdx.x * blockDim.x + threadIdx.x;
    int total = K * HDIM;
    if (t >= total) return;
    int e    = t & 15;
    int lane = (t >> 4) & 31;
    int r    = t >> 9;
    int kchunks = K >> 5;
    int kc    = r % kchunks;
    int ntile = r / kchunks;
    int n  = ntile * 16 + (lane & 15);
    int kb = (lane < 16) ? 0 : 8;
    int k  = kc * 32 + kb + (e < 8 ? e : e + 8);
    Wp[t] = f2bf(W[k * HDIM + n]);
}

// ---------------- fused GEMM: Out = act(A @ W + b) [+ graph_hidden[batch]] ----------------
// one wave computes a 16(M) x 64(N) strip via 4x v_wmma_f32_16x16x32_bf16 per K-step
template <int DO_RELU, int DO_RES>
__global__ void __launch_bounds__(256) k_gemm_bf16(
    const float* __restrict__ A, const __bf16* __restrict__ Wp,
    const float* __restrict__ bias, const float* __restrict__ ghid,
    const int* __restrict__ batch, float* __restrict__ Out,
    int K, int nMt)
{
    const int lane = threadIdx.x & 31;
    const int wave = threadIdx.x >> 5;
    const int mtile = blockIdx.x * 8 + wave;
    if (mtile >= nMt) return;                  // uniform per wave -> EXEC all-ones inside
    const int ngrp = blockIdx.y;               // 4 groups of 64 columns
    const int kchunks = K >> 5;
    const int mrow = lane & 15;
    const int kb = (lane < 16) ? 0 : 8;
    const float* arow = A + (long long)(mtile * 16 + mrow) * K;

    v8f c[4];
    c[0] = (v8f){0,0,0,0,0,0,0,0}; c[1] = c[0]; c[2] = c[0]; c[3] = c[0];

    for (int kc = 0; kc < kchunks; ++kc) {
        const int k0 = kc * 32;
        const float4* pa = reinterpret_cast<const float4*>(arow + k0 + kb);
        float4 a0 = pa[0], a1 = pa[1];
        const float4* pb = reinterpret_cast<const float4*>(arow + k0 + kb + 16);
        float4 a2 = pb[0], a3 = pb[1];

        v16bf av;
        v2bf p;
        p = f2bf_pk(a0.x, a0.y); av[0]  = p[0]; av[1]  = p[1];
        p = f2bf_pk(a0.z, a0.w); av[2]  = p[0]; av[3]  = p[1];
        p = f2bf_pk(a1.x, a1.y); av[4]  = p[0]; av[5]  = p[1];
        p = f2bf_pk(a1.z, a1.w); av[6]  = p[0]; av[7]  = p[1];
        p = f2bf_pk(a2.x, a2.y); av[8]  = p[0]; av[9]  = p[1];
        p = f2bf_pk(a2.z, a2.w); av[10] = p[0]; av[11] = p[1];
        p = f2bf_pk(a3.x, a3.y); av[12] = p[0]; av[13] = p[1];
        p = f2bf_pk(a3.z, a3.w); av[14] = p[0]; av[15] = p[1];

        #pragma unroll
        for (int t = 0; t < 4; ++t) {
            const long long base =
                ((long long)((ngrp * 4 + t) * kchunks + kc) * 32 + lane) * 16;
            v16bf bv = *reinterpret_cast<const v16bf*>(Wp + base);
            c[t] = __builtin_amdgcn_wmma_f32_16x16x32_bf16(
                       false, av, false, bv, (short)0, c[t], false, false);
        }
    }

    const int mbase = mtile * 16 + ((lane < 16) ? 0 : 8);
    int bidx[8];
    #pragma unroll
    for (int v = 0; v < 8; ++v) bidx[v] = DO_RES ? batch[mbase + v] : 0;

    #pragma unroll
    for (int t = 0; t < 4; ++t) {
        const int ncol = (ngrp * 4 + t) * 16 + (lane & 15);
        const float bb = bias[ncol];
        #pragma unroll
        for (int v = 0; v < 8; ++v) {
            float val = c[t][v] + bb;
            if (DO_RELU) val = fmaxf(val, 0.0f);
            if (DO_RES)  val += ghid[(long long)bidx[v] * HDIM + ncol];
            Out[(long long)(mbase + v) * HDIM + ncol] = val;
        }
    }
}

// ---------------- pooling + head ----------------
__global__ void k_zero(float* p, int n) {
    int t = blockIdx.x * blockDim.x + threadIdx.x;
    if (t < n) p[t] = 0.0f;
}
__global__ void k_count(float* cnt, const int* __restrict__ batch, int nN) {
    int t = blockIdx.x * blockDim.x + threadIdx.x;
    if (t < nN) unsafeAtomicAdd(&cnt[batch[t]], 1.0f);
}
__global__ void k_poolsum(const float* __restrict__ h, float* __restrict__ pool,
                          const int* __restrict__ batch, long long total) {
    long long t = (long long)blockIdx.x * blockDim.x + threadIdx.x;
    if (t >= total) return;
    int i = (int)(t >> 6);                     // HDIM/4 == 64 chunks per node
    int c = (int)(t & 63);
    float4 v = reinterpret_cast<const float4*>(h)[t];
    float* p = pool + (long long)batch[i] * HDIM + c * 4;
    unsafeAtomicAdd(p + 0, v.x);
    unsafeAtomicAdd(p + 1, v.y);
    unsafeAtomicAdd(p + 2, v.z);
    unsafeAtomicAdd(p + 3, v.w);
}
__global__ void k_gm(const float* __restrict__ pool, const float* __restrict__ cnt,
                     const float* __restrict__ ghid, float* __restrict__ gm,
                     float* __restrict__ outgm, int total) {
    int t = blockIdx.x * blockDim.x + threadIdx.x;
    if (t >= total) return;
    int g = t >> 8;                            // HDIM == 256
    float val = pool[t] / fmaxf(cnt[g], 1.0f) + ghid[t];
    gm[t] = val;
    outgm[t] = val;
}
__global__ void k_head(const float* __restrict__ gm, const float* __restrict__ Wlin,
                       const float* __restrict__ blin, float* __restrict__ y,
                       int G, int C) {
    int t = blockIdx.x * blockDim.x + threadIdx.x;
    if (t >= G * C) return;
    int g = t / C, c = t % C;
    float acc = blin[c];
    for (int n = 0; n < HDIM; ++n) acc += gm[g * HDIM + n] * Wlin[n * C + c];
    y[t] = acc;
}

static inline unsigned int cdivu(long long a, long long b) {
    return (unsigned int)((a + b - 1) / b);
}

extern "C" void kernel_launch(void* const* d_in, const int* in_sizes, int n_in,
                              void* d_out, int out_size, void* d_ws, size_t ws_size,
                              hipStream_t stream) {
    const float* x     = (const float*)d_in[0];
    const int*   eidx  = (const int*)  d_in[1];
    const int*   batch = (const int*)  d_in[2];
    const float* ghid  = (const float*)d_in[3];
    const float* W0    = (const float*)d_in[4];
    const float* b0    = (const float*)d_in[5];
    const float* Ws    = (const float*)d_in[6];
    const float* bs    = (const float*)d_in[7];
    const float* Wlin  = (const float*)d_in[8];
    const float* blin  = (const float*)d_in[9];

    const int nN = in_sizes[2];
    const int nE = in_sizes[1] / 2;
    const int F  = in_sizes[0] / nN;                 // 128
    const int G  = in_sizes[3] / HDIM;               // 128
    const int C  = in_sizes[9];                      // 10
    const int L  = in_sizes[6] / (HDIM * HDIM);      // 3
    const int nMt = nN / 16;                         // 3125 exactly
    const int* srcp = eidx;
    const int* dstp = eidx + nE;

    // ---- workspace carve-out (all offsets 256B aligned) ----
    char* base = (char*)d_ws;
    size_t off = 0;
    float* inv = (float*)(base + off);  off += (((size_t)nN * 4) + 255) & ~(size_t)255;
    float* bufA = (float*)(base + off); off += (size_t)nN * HDIM * 4;
    float* bufB = (float*)(base + off); off += (size_t)nN * HDIM * 4;
    __bf16* wpack = (__bf16*)(base + off);
    off += ((((size_t)F * HDIM + (size_t)L * HDIM * HDIM) * 2) + 255) & ~(size_t)255;
    float* pool = (float*)(base + off); off += (size_t)G * HDIM * 4;
    float* cnt  = (float*)(base + off); off += (((size_t)G * 4) + 255) & ~(size_t)255;
    float* gm   = (float*)(base + off); off += (size_t)G * HDIM * 4;
    (void)ws_size; (void)n_in; (void)out_size;

    // ---- normalization coefficients ----
    k_fill1 <<<cdivu(nN, 256), 256, 0, stream>>>(inv, nN);
    k_degedge<<<cdivu(nE, 256), 256, 0, stream>>>(inv, dstp, nE);
    k_rsqrt <<<cdivu(nN, 256), 256, 0, stream>>>(inv, nN);

    // ---- pre-pack weights to bf16 WMMA-B layout ----
    k_packw<<<cdivu((long long)F * HDIM, 256), 256, 0, stream>>>(W0, wpack, F);
    for (int l = 0; l < L; ++l)
        k_packw<<<cdivu((long long)HDIM * HDIM, 256), 256, 0, stream>>>(
            Ws + (size_t)l * HDIM * HDIM, wpack + (size_t)F * HDIM + (size_t)l * HDIM * HDIM, HDIM);

    dim3 ggrid(cdivu(nMt, 8), 4);

    // ---- layer 0: agg(x) then GEMM(W0) ; epilogue: +residual, no relu ----
    {
        int D4 = F / 4;
        long long tSelf = (long long)nN * D4;
        long long tEdge = (long long)nE * D4;
        k_aggself<<<cdivu(tSelf, 256), 256, 0, stream>>>(x, bufA, inv, D4, tSelf);
        k_aggedge<<<cdivu(tEdge, 256), 256, 0, stream>>>(x, bufA, inv, srcp, dstp, D4, tEdge);
        k_gemm_bf16<0, 1><<<ggrid, 256, 0, stream>>>(bufA, wpack, b0, ghid, batch, bufB,
                                                     F, nMt);
    }

    // ---- layers 1..L: agg then GEMM(Ws[l]) ; relu; +residual except last ----
    for (int l = 0; l < L; ++l) {
        int D4 = HDIM / 4;
        long long tSelf = (long long)nN * D4;
        long long tEdge = (long long)nE * D4;
        k_aggself<<<cdivu(tSelf, 256), 256, 0, stream>>>(bufB, bufA, inv, D4, tSelf);
        k_aggedge<<<cdivu(tEdge, 256), 256, 0, stream>>>(bufB, bufA, inv, srcp, dstp, D4, tEdge);
        const __bf16* wp = wpack + (size_t)F * HDIM + (size_t)l * HDIM * HDIM;
        const float* bp = bs + (size_t)l * HDIM;
        if (l < L - 1)
            k_gemm_bf16<1, 1><<<ggrid, 256, 0, stream>>>(bufA, wp, bp, ghid, batch, bufB,
                                                         HDIM, nMt);
        else
            k_gemm_bf16<1, 0><<<ggrid, 256, 0, stream>>>(bufA, wp, bp, ghid, batch, bufB,
                                                         HDIM, nMt);
    }

    // ---- mean pool per graph + residual, then linear head ----
    k_zero <<<cdivu(G * HDIM, 256), 256, 0, stream>>>(pool, G * HDIM);
    k_zero <<<cdivu(G, 256), 256, 0, stream>>>(cnt, G);
    k_count<<<cdivu(nN, 256), 256, 0, stream>>>(cnt, batch, nN);
    long long tPool = (long long)nN * (HDIM / 4);
    k_poolsum<<<cdivu(tPool, 256), 256, 0, stream>>>(bufB, pool, batch, tPool);

    float* outY  = (float*)d_out;            // [G*C] first
    float* outGM = (float*)d_out + G * C;    // then [G*HDIM]
    k_gm  <<<cdivu(G * HDIM, 256), 256, 0, stream>>>(pool, cnt, ghid, gm, outGM, G * HDIM);
    k_head<<<cdivu(G * C, 256), 256, 0, stream>>>(gm, Wlin, blin, outY, G, C);
}
